// CrossAttention_3049426780497
// MI455X (gfx1250) — matrix operand order repaired and checked by reading, not compiled, not run on hardware
//
#include <hip/hip_runtime.h>
#include <hip/hip_bf16.h>
#include <math.h>

// ---------------------------------------------------------------------------
// CrossAttention for point clouds on gfx1250 (MI455X).
//   Kernel 1: Q/K/V channel projections as WMMA GEMMs (f16 in, f32 acc).
//   Kernel 2: neighbor gather + positional/score MLPs + softmax + weighted sum
//             (gather/L2-bound, VALU + wave32 shuffles).
// Workspace layout (d_ws): Q | K | V, each B*C*N floats (48 MiB total).
// ---------------------------------------------------------------------------

typedef __attribute__((ext_vector_type(16))) _Float16 v16h;
typedef __attribute__((ext_vector_type(8)))  _Float16 v8h;
typedef __attribute__((ext_vector_type(8)))  float    v8f;

#define CDIM 128
#define NPTS 16384
#define KNBR 16
#define BDIM 2
#define EPSBN 1e-5f

union V16U { v16h v; _Float16 e[16]; };

// ---------------------------------------------------------------------------
// Kernel 1: Y[b,o,n] = sum_c W[o,c] * X[b,c,n] (+ bias[o])
// Block = 256 threads = 8 waves; each wave owns 16 columns (n) and all 128
// output rows (o). 32 v_wmma_f32_16x16x32_f16 per wave.
// ---------------------------------------------------------------------------
__global__ __launch_bounds__(256) void qkv_gemm(
    const float* __restrict__ X,     // [B, C, N]
    const float* __restrict__ W,     // [C, C] row-major (o, c)
    const float* __restrict__ bias,  // [C] or nullptr
    float* __restrict__ Y)           // [B, C, N]
{
  // Stage weights in LDS as f16 (32 KB), cooperative coalesced load.
  __shared__ _Float16 wlds[CDIM * CDIM];
  const int tid = threadIdx.x;
  for (int i = tid; i < CDIM * CDIM; i += 256)
    wlds[i] = (_Float16)W[i];
  __syncthreads();

  const int lane = tid & 31;
  const int wave = tid >> 5;
  const int b    = blockIdx.y;
  const int n0   = (blockIdx.x * 8 + wave) * 16;
  const int ncol = lane & 15;
  const int kofs = lane & 16;   // upper half-wave flag (16)
  const int amsb = kofs >> 1;   // A-frag: upper half-wave holds K +8

  // ---- B fragments: X tile [128 x 16] split into 4 chunks of K=32. ----
  // B layout (K x N = 32 x 16), matching A: lane l (col N=l&15) holds
  // halfs 0..7 -> K = amsb+0..7, halfs 8..15 -> K = amsb+16..23.
  const float* xb = X + (size_t)b * CDIM * NPTS + n0 + ncol;
  v16h bf[4];
  #pragma unroll
  for (int cb = 0; cb < 4; ++cb) {
    V16U u;
    #pragma unroll
    for (int j = 0; j < 16; ++j)
      u.e[j] = (_Float16)xb[(size_t)(cb * 32 + amsb + j + (j & 8)) * NPTS];
    bf[cb] = u.v;
  }

  v8f acc[8];
  #pragma unroll
  for (int ot = 0; ot < 8; ++ot) acc[ot] = (v8f){0.f,0.f,0.f,0.f,0.f,0.f,0.f,0.f};

  // ---- A fragments from LDS; ISA 16-bit A layout (M x K = 16 x 32):   ----
  // lane row M = lane&15; halfs 0..7 -> K = amsb+0..7, halfs 8..15 ->
  // K = amsb+16..23. Both runs are contiguous -> two 16B LDS loads each.
  const int arow = (lane & 15) * CDIM;
  #pragma unroll
  for (int ot = 0; ot < 8; ++ot) {
    const _Float16* wrow = &wlds[ot * 16 * CDIM + arow + amsb];
    #pragma unroll
    for (int cb = 0; cb < 4; ++cb) {
      v8h lo = *(const v8h*)(wrow + cb * 32);
      v8h hi = *(const v8h*)(wrow + cb * 32 + 16);
      v16h a = __builtin_shufflevector(lo, hi,
                 0,1,2,3,4,5,6,7,8,9,10,11,12,13,14,15);
      acc[ot] = __builtin_amdgcn_wmma_f32_16x16x32_f16(
          /*neg_a=*/false, a, /*neg_b=*/false, bf[cb],
          /*c_mod=*/(short)0, acc[ot], /*reuse_a=*/false, /*reuse_b=*/false);
    }
  }

  // ---- Store: f32 C/D layout -> VGPR v holds (M = v + 8*(lane>=16)). ----
  float* yb = Y + (size_t)b * CDIM * NPTS + n0 + ncol;
  #pragma unroll
  for (int ot = 0; ot < 8; ++ot) {
    #pragma unroll
    for (int v = 0; v < 8; ++v) {
      const int o = ot * 16 + v + amsb;
      float r = acc[ot][v];
      if (bias) r += bias[o];
      yb[(size_t)o * NPTS] = r;
    }
  }
}

// ---------------------------------------------------------------------------
// Kernel 2: per-point neighbor attention. One wave = 2 points; within a
// half-wave lane k owns neighbor k for scoring, then 8 channels (k + 16*j)
// for the attn-weighted V reduction. Softmax via width-16 shfl_xor.
// ---------------------------------------------------------------------------
__global__ __launch_bounds__(256) void attn_fuse(
    const float* __restrict__ coord, const int* __restrict__ nbr,
    const float* __restrict__ Q, const float* __restrict__ Kf,
    const float* __restrict__ V,
    const float* __restrict__ pw1, const float* __restrict__ pb1,
    const float* __restrict__ pg,  const float* __restrict__ pbeta,
    const float* __restrict__ pmu, const float* __restrict__ pvar,
    const float* __restrict__ pw2, const float* __restrict__ pb2,
    const float* __restrict__ sw1, const float* __restrict__ sb1,
    const float* __restrict__ sg,  const float* __restrict__ sbeta,
    const float* __restrict__ smu, const float* __restrict__ svar,
    const float* __restrict__ sw2, const float* __restrict__ sb2,
    float* __restrict__ out)
{
  const int gtid = blockIdx.x * blockDim.x + threadIdx.x;
  const int lane = gtid & 31;
  const int wid  = gtid >> 5;
  const int sub  = lane >> 4;        // half-wave id
  const int k    = lane & 15;        // neighbor slot
  const int p    = wid * 2 + sub;    // global point id
  if (p >= BDIM * NPTS) return;
  const int b = p >> 14;             // NPTS = 2^14
  const int n = p & (NPTS - 1);

  const int idx = nbr[(size_t)(b * NPTS + n) * KNBR + k];

  // ---- positional branch: g = exp(-2*diff^2); 3->3 conv + BN + relu; 3->1
  const float* cc = coord + (size_t)(b * NPTS + n)   * 3;
  const float* cn = coord + (size_t)(b * NPTS + idx) * 3;
  float gv[3];
  #pragma unroll
  for (int d = 0; d < 3; ++d) {
    const float df = cn[d] - cc[d];
    gv[d] = __expf(-2.f * df * df);
  }
  float h[3];
  #pragma unroll
  for (int o = 0; o < 3; ++o) {
    float t = pw1[o*3+0]*gv[0] + pw1[o*3+1]*gv[1] + pw1[o*3+2]*gv[2] + pb1[o];
    const float inv = pg[o] * rsqrtf(pvar[o] + EPSBN);
    t = t * inv + (pbeta[o] - pmu[o] * inv);
    h[o] = fmaxf(t, 0.f);
  }
  const float s_coord = pw2[0]*h[0] + pw2[1]*h[1] + pw2[2]*h[2] + pb2[0];

  // ---- feature branch: dot(Q[:,n], Kf[:,idx]) / sqrt(C)
  const float* qp = Q  + (size_t)b * CDIM * NPTS + n;
  const float* kp = Kf + (size_t)b * CDIM * NPTS + idx;
  float dot = 0.f;
  #pragma unroll 8
  for (int c = 0; c < CDIM; ++c)
    dot = fmaf(qp[(size_t)c * NPTS], kp[(size_t)c * NPTS], dot);
  const float s_feat = dot * 0.08838834764831845f;   // 1/sqrt(128)

  // ---- score MLP 2->2 (BN+relu) -> 1
  float z[2];
  #pragma unroll
  for (int o = 0; o < 2; ++o) {
    float t = sw1[o*2+0] * s_coord + sw1[o*2+1] * s_feat + sb1[o];
    const float inv = sg[o] * rsqrtf(svar[o] + EPSBN);
    t = t * inv + (sbeta[o] - smu[o] * inv);
    z[o] = fmaxf(t, 0.f);
  }
  float s = sw2[0]*z[0] + sw2[1]*z[1] + sb2[0];

  // ---- softmax over the 16 neighbors of this half-wave
  float m = s;
  #pragma unroll
  for (int off = 8; off; off >>= 1) m = fmaxf(m, __shfl_xor(m, off, 16));
  const float e = __expf(s - m);
  float sum = e;
  #pragma unroll
  for (int off = 8; off; off >>= 1) sum += __shfl_xor(sum, off, 16);
  const float attn = e / sum;

  // ---- broadcast (attn, idx) of all 16 neighbors within the half-wave
  float aall[KNBR];
  int   iall[KNBR];
  const int base = lane & 16;
  #pragma unroll
  for (int kk = 0; kk < KNBR; ++kk) {
    aall[kk] = __shfl(attn, base + kk, 32);
    iall[kk] = __shfl(idx,  base + kk, 32);
  }

  // ---- weighted V sum: lane covers channels c = k + 16*j, j = 0..7
  const float* vb = V   + (size_t)b * CDIM * NPTS;
  float*       ob = out + (size_t)b * CDIM * NPTS + n;
  #pragma unroll
  for (int j = 0; j < 8; ++j) {
    const int c = k + 16 * j;
    const float* vrow = vb + (size_t)c * NPTS;
    float acc = 0.f;
    #pragma unroll
    for (int kk = 0; kk < KNBR; ++kk)
      acc = fmaf(aall[kk], vrow[iall[kk]], acc);
    ob[(size_t)c * NPTS] = acc;
  }
}

// ---------------------------------------------------------------------------
extern "C" void kernel_launch(void* const* d_in, const int* in_sizes, int n_in,
                              void* d_out, int out_size, void* d_ws, size_t ws_size,
                              hipStream_t stream) {
  const float* coord = (const float*)d_in[0];
  const float* spa   = (const float*)d_in[1];
  const float* spe   = (const float*)d_in[2];
  const int*   nbr   = (const int*)  d_in[3];
  const float* wq    = (const float*)d_in[4];
  const float* wk    = (const float*)d_in[5];
  const float* wv_w  = (const float*)d_in[6];
  const float* wv_b  = (const float*)d_in[7];
  const float* pw1   = (const float*)d_in[8];
  const float* pb1   = (const float*)d_in[9];
  const float* pg    = (const float*)d_in[10];
  const float* pbeta = (const float*)d_in[11];
  const float* pmu   = (const float*)d_in[12];
  const float* pvar  = (const float*)d_in[13];
  const float* pw2   = (const float*)d_in[14];
  const float* pb2   = (const float*)d_in[15];
  const float* sw1   = (const float*)d_in[16];
  const float* sb1   = (const float*)d_in[17];
  const float* sg    = (const float*)d_in[18];
  const float* sbeta = (const float*)d_in[19];
  const float* smu   = (const float*)d_in[20];
  const float* svar  = (const float*)d_in[21];
  const float* sw2   = (const float*)d_in[22];
  const float* sb2   = (const float*)d_in[23];

  const size_t per = (size_t)BDIM * CDIM * NPTS;  // 4 Mi floats
  float* Qb = (float*)d_ws;
  float* Kb = Qb + per;
  float* Vb = Kb + per;

  // GEMM grid: x covers N in 128-column blocks (8 waves x 16 cols), y = batch.
  dim3 gg(NPTS / 128, BDIM, 1);
  qkv_gemm<<<gg, 256, 0, stream>>>(spa, wq,   nullptr, Qb);
  qkv_gemm<<<gg, 256, 0, stream>>>(spe, wk,   nullptr, Kb);
  qkv_gemm<<<gg, 256, 0, stream>>>(spe, wv_w, wv_b,    Vb);

  // Attention grid: 2 points per wave, 8 waves per block.
  const int total_waves = BDIM * NPTS / 2;          // 16384
  const int blocks = total_waves * 32 / 256;        // 2048
  attn_fuse<<<blocks, 256, 0, stream>>>(coord, nbr, Qb, Kb, Vb,
      pw1, pb1, pg, pbeta, pmu, pvar, pw2, pb2,
      sw1, sb1, sg, sbeta, smu, svar, sw2, sb2, (float*)d_out);
}
